// AttentionBlockE3_63806034149758
// MI455X (gfx1250) — compile-verified, matrix operand order checked by reading, not run on hardware
//
#include <hip/hip_runtime.h>

typedef __attribute__((ext_vector_type(16))) _Float16 v16h;
typedef __attribute__((ext_vector_type(8)))  _Float16 v8h;
typedef __attribute__((ext_vector_type(2)))  _Float16 v2h;
typedef __attribute__((ext_vector_type(8)))  float    v8f;

#define RS2  0.70710678118654752f   // 1/sqrt(2)
#define RS3  0.57735026918962576f   // 1/sqrt(3)
#define RSHD 0.17677669529663688f   // 1/sqrt(32) (HEAD_DIM)
#define RS8  0.35355339059327378f   // 1/sqrt(8)

// planar index t=p*64+k  ->  x-order feature index d
__device__ __forceinline__ int plan2d(int t) {
  int p = t >> 6, k = t & 63;
  return (p == 0) ? k : (64 + 3 * k + (p - 1));
}

// 16-half WMMA fragment = two contiguous 8-half (16B) runs at p and p+16.
// Works for both A (from planar f16 LDS rows) and B (from transposed f16 weights).
__device__ __forceinline__ v16h load_frag16(const _Float16* __restrict__ p) {
  v8h lo = *(const v8h*)p;
  v8h hi = *(const v8h*)(p + 16);
  v16h r;
#pragma unroll
  for (int j = 0; j < 8; ++j) { r[j] = lo[j]; r[j + 8] = hi[j]; }
  return r;
}

__device__ __forceinline__ float sigmoidf_(float x) { return 1.f / (1.f + __expf(-x)); }
__device__ __forceinline__ float siluf_(float x) { return x * sigmoidf_(x); }
__device__ __forceinline__ unsigned f2key(float f) {
  unsigned b = __float_as_uint(f);
  return (b & 0x80000000u) ? ~b : (b | 0x80000000u);
}
__device__ __forceinline__ float key2f(unsigned k) {
  unsigned b = (k & 0x80000000u) ? (k ^ 0x80000000u) : ~k;
  return __uint_as_float(b);
}

// ---------------- K1: convert/transpose weights to f16 ----------------
__global__ void convert_weights(const float* __restrict__ W0, const float* __restrict__ W1,
                                const float* __restrict__ mw0, const float* __restrict__ mw1,
                                const float* __restrict__ mw2, const float* __restrict__ mw3,
                                _Float16* W0t, _Float16* W1t, _Float16* mw0t,
                                _Float16* mw1t, _Float16* mw2t, _Float16* mw3t) {
  int id = blockIdx.x * 256 + threadIdx.x;
  if (id < 24576) {                       // W0t[i][n][k] = W0[i][k][n]
    int i = id >> 12, rem = id & 4095, n = rem >> 6, k = rem & 63;
    W0t[id] = (_Float16)W0[i * 4096 + k * 64 + n];
  } else if (id < 49152) {
    int j = id - 24576; int i = j >> 12, rem = j & 4095, n = rem >> 6, k = rem & 63;
    W1t[j] = (_Float16)W1[i * 4096 + k * 64 + n];
  } else if (id < 51200) {                // mw0t: 64 n x 32 k, K zero-padded 8->32
    int j = id - 49152; int n = j >> 5, k = j & 31;
    mw0t[j] = (_Float16)((k < 8) ? mw0[k * 64 + n] : 0.f);
  } else if (id < 55296) {
    int j = id - 51200; int n = j >> 6, k = j & 63;
    mw1t[j] = (_Float16)mw1[k * 64 + n];
  } else if (id < 59392) {
    int j = id - 55296; int n = j >> 6, k = j & 63;
    mw2t[j] = (_Float16)mw2[k * 64 + n];
  } else if (id < 75776) {                // mw3t: 256 n x 64 k
    int j = id - 59392; int n = j >> 6, k = j & 63;
    mw3t[j] = (_Float16)mw3[k * 256 + n];
  }
}

__global__ void init_accum(unsigned* nmaxkey, float* den, float* agg, int N_) {
  int id = blockIdx.x * 256 + threadIdx.x;
  if (id < N_ * 8) { nmaxkey[id] = 0u; den[id] = 0.f; }
  if (id < N_ * 256) agg[id] = 0.f;
}

// ---------- irrep linear on a planar-f16 16-row tile ----------
// sh_in : [4 parts][16 rows][64 k] f16 ; sh_out same layout.
// task = p*4+nt ; two WMMA k-steps of 32.
__device__ __forceinline__ void irrep_tile_wmma_h(const _Float16* __restrict__ sh_in,
                                                  const _Float16* __restrict__ W0t,
                                                  const _Float16* __restrict__ W1t,
                                                  int wave, int lane,
                                                  _Float16* __restrict__ sh_out,
                                                  bool accumulate) {
  int g = (lane >> 4) & 1;
  int l15 = lane & 15;
  for (int task = wave; task < 16; task += 8) {
    int p = task >> 2, nt = task & 3;
    const _Float16* Wt = (p == 0) ? W0t : W1t;
    const _Float16* arow = sh_in + (p * 16 + l15) * 64;
    const _Float16* brow = Wt + (nt * 16 + l15) * 64;
    v8f acc = {};
#pragma unroll
    for (int k0 = 0; k0 < 64; k0 += 32) {
      v16h a = load_frag16(arow + k0 + g * 8);
      v16h b = load_frag16(brow + k0 + g * 8);
      acc = __builtin_amdgcn_wmma_f32_16x16x32_f16(false, a, false, b, (short)0, acc, false, false);
    }
    int n = nt * 16 + l15;
    _Float16* ocol = sh_out + p * 16 * 64 + n;
#pragma unroll
    for (int r = 0; r < 8; ++r) {
      int m = r + g * 8;
      if (accumulate) ocol[m * 64] = (_Float16)((float)ocol[m * 64] + acc[r] * 0.125f);
      else            ocol[m * 64] = (_Float16)(acc[r] * 0.125f);
    }
  }
}

// ---------------- K2: four node irrep linears -> planar f16 rows ----------------
__global__ void node_linears(const float* __restrict__ X, int nrows,
                             const _Float16* __restrict__ W0t_all,
                             const _Float16* __restrict__ W1t_all,
                             _Float16* lin0h, _Float16* lin1h, _Float16* kqh, _Float16* sch) {
  __shared__ __align__(16) _Float16 sh_x[4 * 16 * 64];
  __shared__ __align__(16) _Float16 sh_o[4 * 16 * 64];
  int row0 = blockIdx.x * 16;
  int t = threadIdx.x, wave = t >> 5, lane = t & 31;
  const int widx[4] = {0, 1, 3, 4};
  _Float16* outs[4] = {lin0h, lin1h, kqh, sch};
  int sel = blockIdx.y;
  const _Float16* W0t = W0t_all + widx[sel] * 4096;
  const _Float16* W1t = W1t_all + widx[sel] * 4096;
  _Float16* out = outs[sel];

  int d = plan2d(t), p = t >> 6, k = t & 63;
  for (int m = 0; m < 16; ++m) {
    int r = row0 + m;
    sh_x[(p * 16 + m) * 64 + k] = (_Float16)((r < nrows) ? X[(size_t)r * 256 + d] : 0.f);
  }
  __syncthreads();
  irrep_tile_wmma_h(sh_x, W0t, W1t, wave, lane, sh_o, false);
  __syncthreads();
  // coalesced dword copy-out (planar global layout)
  for (int it = 0; it < 8; ++it) {
    int m = it * 2 + (t >> 7);
    int td = t & 127, pp = td >> 5, w2 = td & 31;
    int r = row0 + m;
    if (r < nrows) {
      unsigned v = *(const unsigned*)&sh_o[(pp * 16 + m) * 64 + 2 * w2];
      ((unsigned*)out)[(size_t)r * 128 + td] = v;
    }
  }
}

// ---------------- K3: fused edge kernel ----------------
__global__ void edge_kernel(const float* __restrict__ edge_feats,
                            const float* __restrict__ edge_attrs,
                            const float* __restrict__ edge_embed,
                            const float* __restrict__ edge_length,
                            const int* __restrict__ edge_index, int E_,
                            const _Float16* __restrict__ lin0h, const _Float16* __restrict__ lin1h,
                            const _Float16* __restrict__ kqh,
                            const _Float16* __restrict__ W0t2, const _Float16* __restrict__ W1t2,
                            const _Float16* __restrict__ mw0t, const _Float16* __restrict__ mw1t,
                            const _Float16* __restrict__ mw2t, const _Float16* __restrict__ mw3t,
                            float* __restrict__ logits, _Float16* __restrict__ valh,
                            unsigned* __restrict__ nmaxkey) {
  __shared__ __align__(16) float    sh_raw[16 * 256];   // async-staged raw edge_feats
  __shared__ __align__(16) _Float16 sh_ef[4 * 16 * 64]; // planar f16 edge_feats
  __shared__ __align__(16) _Float16 sh_x[4 * 16 * 64];  // gathered x -> value (in place)
  __shared__ __align__(16) float    sh_w[16 * 256];     // radial output w
  __shared__ __align__(16) _Float16 sh_e[16 * 32];      // padded radial input
  __shared__ __align__(16) _Float16 sh_h[2][16 * 64];   // radial hidden ping-pong
  __shared__ int   snd[16], rcv[16];
  __shared__ float Y0s[16], Y1s[16][3], cut[16];

  int e0 = blockIdx.x * 16;
  int t = threadIdx.x, wave = t >> 5, lane = t & 31;

  // ---- async HBM -> LDS staging of the edge_feats tile (ASYNCcnt path) ----
  {
    const char* gbase = (const char*)(edge_feats + (size_t)e0 * 256);
    unsigned lbase = (unsigned)(size_t)(void*)sh_raw;
#pragma unroll
    for (int it = 0; it < 4; ++it) {
      unsigned loff = lbase + it * 4096 + t * 16;
      const char* g = gbase + it * 4096 + t * 16;
      asm volatile("global_load_async_to_lds_b128 %0, %1, off" :: "v"(loff), "v"(g) : "memory");
    }
  }
  if (t < 16) {
    int e = e0 + t; int ec = (e < E_) ? e : (E_ - 1);
    snd[t] = edge_index[ec];
    rcv[t] = edge_index[E_ + ec];
    Y0s[t] = edge_attrs[ec * 4 + 0];
    Y1s[t][0] = edge_attrs[ec * 4 + 1];
    Y1s[t][1] = edge_attrs[ec * 4 + 2];
    Y1s[t][2] = edge_attrs[ec * 4 + 3];
    float xl = 10.f * (1.f - edge_length[ec]);
    cut[t] = (xl > 0.f) ? __expf(-1.f / xl) : 0.f;
  }
  { // padded radial input, 1/sqrt(8) folded in
    int m = t >> 5, k = t & 31;
#pragma unroll
    for (int half = 0; half < 2; ++half, m += 8) {
      int ec = (e0 + m < E_) ? e0 + m : E_ - 1;
      sh_e[m * 32 + k] = (_Float16)((k < 8) ? edge_embed[(size_t)ec * 8 + k] * RS8 : 0.f);
    }
  }
  asm volatile("s_wait_asynccnt 0" ::: "memory");
  __syncthreads();

  // convert raw tile -> planar f16 ; gather lin0[snd]+lin1[rcv] (planar f16 rows)
  {
    int p = t >> 6, k = t & 63, d = plan2d(t);
    for (int m = 0; m < 16; ++m)
      sh_ef[(p * 16 + m) * 64 + k] = (_Float16)sh_raw[m * 256 + d];
  }
#pragma unroll
  for (int it = 0; it < 8; ++it) {
    int m = it * 2 + (t >> 7);
    int td = t & 127, pp = td >> 5, w2 = td & 31;
    unsigned ua = ((const unsigned*)(lin0h + (size_t)snd[m] * 256))[td];
    unsigned ub = ((const unsigned*)(lin1h + (size_t)rcv[m] * 256))[td];
    v2h ha = __builtin_bit_cast(v2h, ua);
    v2h hb = __builtin_bit_cast(v2h, ub);
    v2h hs;
    hs[0] = (_Float16)((float)ha[0] + (float)hb[0]);
    hs[1] = (_Float16)((float)ha[1] + (float)hb[1]);
    *(v2h*)&sh_x[(pp * 16 + m) * 64 + 2 * w2] = hs;
  }
  __syncthreads();

  // fused edge irrep-linear: sh_x += edge_feats @ W[2]
  irrep_tile_wmma_h(sh_ef, W0t2, W1t2, wave, lane, sh_x, true);
  __syncthreads();

  // ---- radial MLP (all WMMA) ----
  int g = (lane >> 4) & 1, l15 = lane & 15;
  if (wave < 4) {  // h1 = silu(e @ mw0 / sqrt8), K padded to 32
    v8f acc = {};
    v16h a = load_frag16(sh_e + l15 * 32 + g * 8);
    v16h b = load_frag16(mw0t + (wave * 16 + l15) * 32 + g * 8);
    acc = __builtin_amdgcn_wmma_f32_16x16x32_f16(false, a, false, b, (short)0, acc, false, false);
    int n = wave * 16 + l15;
#pragma unroll
    for (int r = 0; r < 8; ++r) sh_h[0][(r + g * 8) * 64 + n] = (_Float16)siluf_(acc[r]);
  }
  __syncthreads();
#pragma unroll
  for (int layer = 0; layer < 2; ++layer) {  // h2, h3
    const _Float16* Wt = (layer == 0) ? mw1t : mw2t;
    if (wave < 4) {
      v8f acc = {};
#pragma unroll
      for (int k0 = 0; k0 < 64; k0 += 32) {
        v16h a = load_frag16(sh_h[layer & 1] + l15 * 64 + k0 + g * 8);
        v16h b = load_frag16(Wt + (wave * 16 + l15) * 64 + k0 + g * 8);
        acc = __builtin_amdgcn_wmma_f32_16x16x32_f16(false, a, false, b, (short)0, acc, false, false);
      }
      int n = wave * 16 + l15;
#pragma unroll
      for (int r = 0; r < 8; ++r)
        sh_h[1 - (layer & 1)][(r + g * 8) * 64 + n] = (_Float16)siluf_(acc[r] * 0.125f);
    }
    __syncthreads();
  }
  for (int task = wave; task < 16; task += 8) {  // w = h3 @ mw3 / 8
    v8f acc = {};
#pragma unroll
    for (int k0 = 0; k0 < 64; k0 += 32) {
      v16h a = load_frag16(sh_h[0] + l15 * 64 + k0 + g * 8);
      v16h b = load_frag16(mw3t + (task * 16 + l15) * 64 + k0 + g * 8);
      acc = __builtin_amdgcn_wmma_f32_16x16x32_f16(false, a, false, b, (short)0, acc, false, false);
    }
    int n = task * 16 + l15;
#pragma unroll
    for (int r = 0; r < 8; ++r) sh_w[(r + g * 8) * 256 + n] = acc[r] * 0.125f;
  }
  __syncthreads();

  // ---- tensor-product value, in place over planar sh_x ----
#pragma unroll
  for (int it = 0; it < 4; ++it) {
    int m = (t >> 6) + 4 * it;
    int u = t & 63;
    float s  = (float)sh_x[(0 * 16 + m) * 64 + u];
    float v0 = (float)sh_x[(1 * 16 + m) * 64 + u];
    float v1 = (float)sh_x[(2 * 16 + m) * 64 + u];
    float v2 = (float)sh_x[(3 * 16 + m) * 64 + u];
    float w1 = sh_w[m * 256 + u];
    float w2 = sh_w[m * 256 + 64 + u];
    float w3 = sh_w[m * 256 + 128 + u];
    float w4 = sh_w[m * 256 + 192 + u];
    float y0 = Y0s[m], ya = Y1s[m][0], yb = Y1s[m][1], yc = Y1s[m][2];
    float dv = v0 * ya + v1 * yb + v2 * yc;
    sh_x[(0 * 16 + m) * 64 + u] = (_Float16)((w1 * s * y0 + w4 * dv * RS3) * RS2);
    sh_x[(1 * 16 + m) * 64 + u] = (_Float16)((w2 * s * ya + w3 * v0 * y0) * RS2);
    sh_x[(2 * 16 + m) * 64 + u] = (_Float16)((w2 * s * yb + w3 * v1 * y0) * RS2);
    sh_x[(3 * 16 + m) * 64 + u] = (_Float16)((w2 * s * yc + w3 * v2 * y0) * RS2);
  }
  __syncthreads();

  // value out (planar f16, coalesced dwords)
#pragma unroll
  for (int it = 0; it < 8; ++it) {
    int m = it * 2 + (t >> 7);
    int td = t & 127, pp = td >> 5, w2 = td & 31;
    int e = e0 + m;
    if (e < E_) {
      unsigned v = *(const unsigned*)&sh_x[(pp * 16 + m) * 64 + 2 * w2];
      ((unsigned*)valh)[(size_t)e * 128 + td] = v;
    }
  }
  // per-head logits + segment max (planar: head h = 4 contiguous 8-half runs)
  if (t < 128) {
    int r = t >> 3, h = t & 7;
    int e = e0 + r;
    if (e < E_) {
      const _Float16* krow = kqh + (size_t)snd[r] * 256;
      const _Float16* qrow = kqh + (size_t)rcv[r] * 256;
      float dot = 0.f;
#pragma unroll
      for (int p = 0; p < 4; ++p)
#pragma unroll
        for (int j = 0; j < 8; ++j)
          dot += (float)qrow[p * 64 + h * 8 + j] * (float)krow[p * 64 + h * 8 + j];
      float lg = cut[r] * dot * RSHD;
      logits[(size_t)e * 8 + h] = lg;
      atomicMax(&nmaxkey[(size_t)rcv[r] * 8 + h], f2key(lg));
    }
  }
}

// ---------------- K4: one-pass softmax aggregation ----------------
__global__ void softagg_kernel(const int* __restrict__ edge_index, int E_,
                               const float* __restrict__ logits,
                               const _Float16* __restrict__ valh,
                               const unsigned* __restrict__ nmaxkey,
                               float* __restrict__ den, float* __restrict__ agg) {
  __shared__ float exsh[16][8];
  __shared__ int rcv[16];
  int e0 = blockIdx.x * 16;
  int t = threadIdx.x;
  if (t < 16) {
    int e = e0 + t; int ec = (e < E_) ? e : (E_ - 1);
    rcv[t] = edge_index[E_ + ec];
  }
  __syncthreads();
  if (t < 128) {
    int r = t >> 3, h = t & 7;
    int e = e0 + r;
    float ex = 0.f;
    if (e < E_) {
      float m = key2f(nmaxkey[(size_t)rcv[r] * 8 + h]);
      ex = __expf(logits[(size_t)e * 8 + h] - m);
      atomicAdd(&den[(size_t)rcv[r] * 8 + h], ex);
    }
    exsh[r][h] = ex;
  }
  __syncthreads();
  int d = t;
  int h = (d & 63) >> 3;  // head in planar order
  for (int r = 0; r < 16; ++r) {
    int e = e0 + r;
    if (e >= E_) continue;
    float v = (float)valh[(size_t)e * 256 + d];
    atomicAdd(&agg[(size_t)rcv[r] * 256 + d], exsh[r][h] * v);
  }
}

// ---------------- K5: normalize, final irrep linear, gates, residuals ----------------
__global__ void final_kernel(const float* __restrict__ agg, const float* __restrict__ den,
                             const _Float16* __restrict__ sch,
                             const _Float16* __restrict__ W0t5, const _Float16* __restrict__ W1t5,
                             int nrows, float* __restrict__ out) {
  __shared__ __align__(16) _Float16 sh_f[4 * 16 * 64];
  __shared__ __align__(16) _Float16 sh_r[4 * 16 * 64];
  int row0 = blockIdx.x * 16;
  int t = threadIdx.x, wave = t >> 5, lane = t & 31;
  int h = (t & 63) >> 3;
  for (int m = 0; m < 16; ++m) {
    int n = row0 + m; int nc = (n < nrows) ? n : (nrows - 1);
    float dn = den[(size_t)nc * 8 + h];
    float a  = agg[(size_t)nc * 256 + t];
    sh_f[((t >> 6) * 16 + m) * 64 + (t & 63)] = (_Float16)((dn > 0.f) ? (a / dn) : 0.f);
  }
  __syncthreads();
  irrep_tile_wmma_h(sh_f, W0t5, W1t5, wave, lane, sh_r, false);
  __syncthreads();
  if (t < 128) {
    for (int m = 0; m < 16; ++m) {
      int n = row0 + m;
      if (n >= nrows) continue;
      const _Float16* scrow = sch + (size_t)n * 256;
      if (t < 64) {
        float r = (float)sh_r[(0 * 16 + m) * 64 + t];
        float f = (float)sh_f[(0 * 16 + m) * 64 + t];
        out[(size_t)n * 256 + t] = siluf_(r) + f + (float)scrow[t];
      } else {
        int u = t - 64;
        float r0 = (float)sh_r[(1 * 16 + m) * 64 + u];
        float r1 = (float)sh_r[(2 * 16 + m) * 64 + u];
        float r2 = (float)sh_r[(3 * 16 + m) * 64 + u];
        float vn = sqrtf(r0 * r0 + r1 * r1 + r2 * r2 + 1e-16f);
        float gg = sigmoidf_(vn);  // silu(vn)/vn
#pragma unroll
        for (int c = 0; c < 3; ++c) {
          int d = 64 + 3 * u + c;
          float rv = (c == 0) ? r0 : (c == 1) ? r1 : r2;
          float f = (float)sh_f[((1 + c) * 16 + m) * 64 + u];
          out[(size_t)n * 256 + d] = rv * gg + f + (float)scrow[(1 + c) * 64 + u];
        }
      }
    }
  }
}

// ---------------- host launcher ----------------
extern "C" void kernel_launch(void* const* d_in, const int* in_sizes, int n_in,
                              void* d_out, int out_size, void* d_ws, size_t ws_size,
                              hipStream_t stream) {
  const float* node_feats  = (const float*)d_in[0];
  const float* edge_feats  = (const float*)d_in[1];
  const float* edge_attrs  = (const float*)d_in[2];
  const float* edge_embed  = (const float*)d_in[3];
  const float* edge_length = (const float*)d_in[4];
  const int*   edge_index  = (const int*)d_in[5];
  const float* W0  = (const float*)d_in[6];
  const float* W1  = (const float*)d_in[7];
  const float* mw0 = (const float*)d_in[8];
  const float* mw1 = (const float*)d_in[9];
  const float* mw2 = (const float*)d_in[10];
  const float* mw3 = (const float*)d_in[11];
  float* out = (float*)d_out;

  const int N_ = in_sizes[0] / 256;
  const int E_ = in_sizes[1] / 256;

  char* p = (char*)d_ws;
  _Float16* lin0h = (_Float16*)p; p += (size_t)N_ * 256 * 2;
  _Float16* lin1h = (_Float16*)p; p += (size_t)N_ * 256 * 2;
  _Float16* kqh   = (_Float16*)p; p += (size_t)N_ * 256 * 2;
  _Float16* sch   = (_Float16*)p; p += (size_t)N_ * 256 * 2;
  float* agg  = (float*)p; p += (size_t)N_ * 256 * 4;
  float* logits = (float*)p; p += (size_t)E_ * 8 * 4;
  float* den  = (float*)p; p += (size_t)N_ * 8 * 4;
  unsigned* nmaxkey = (unsigned*)p; p += (size_t)N_ * 8 * 4;
  _Float16* W0t  = (_Float16*)p; p += 24576 * 2;
  _Float16* W1t  = (_Float16*)p; p += 24576 * 2;
  _Float16* mw0t = (_Float16*)p; p += 2048 * 2;
  _Float16* mw1t = (_Float16*)p; p += 4096 * 2;
  _Float16* mw2t = (_Float16*)p; p += 4096 * 2;
  _Float16* mw3t = (_Float16*)p; p += 16384 * 2;
  _Float16* valh = (_Float16*)p; p += (size_t)E_ * 256 * 2;
  (void)ws_size; (void)n_in; (void)out_size;

  const int nTilesN = (N_ + 15) / 16;
  const int nTilesE = (E_ + 15) / 16;

  convert_weights<<<(75776 + 255) / 256, 256, 0, stream>>>(W0, W1, mw0, mw1, mw2, mw3,
                                                           W0t, W1t, mw0t, mw1t, mw2t, mw3t);
  init_accum<<<(N_ * 256 + 255) / 256, 256, 0, stream>>>(nmaxkey, den, agg, N_);
  node_linears<<<dim3(nTilesN, 4), 256, 0, stream>>>(node_feats, N_, W0t, W1t,
                                                     lin0h, lin1h, kqh, sch);
  edge_kernel<<<nTilesE, 256, 0, stream>>>(edge_feats, edge_attrs, edge_embed, edge_length,
                                           edge_index, E_, lin0h, lin1h, kqh,
                                           W0t + 2 * 4096, W1t + 2 * 4096,
                                           mw0t, mw1t, mw2t, mw3t,
                                           logits, valh, nmaxkey);
  softagg_kernel<<<nTilesE, 256, 0, stream>>>(edge_index, E_, logits, valh, nmaxkey, den, agg);
  final_kernel<<<nTilesN, 256, 0, stream>>>(agg, den, sch, W0t + 5 * 4096, W1t + 5 * 4096,
                                            N_, out);
}
